// GNN_node_40888088658269
// MI455X (gfx1250) — compile-verified, matrix operand order checked by reading, not run on hardware
//
#include <hip/hip_runtime.h>

#define EMB 256
#define HID 512

typedef __bf16 v16bf __attribute__((ext_vector_type(16)));
typedef float  v8f   __attribute__((ext_vector_type(8)));
typedef unsigned int u32x4 __attribute__((ext_vector_type(4)));

union Frag {
    u32x4 u[2];
    v16bf v;
};

__device__ __forceinline__ unsigned short f2bf(float f) {
    unsigned int u = __float_as_uint(f);
    unsigned int r = u + 0x7FFFu + ((u >> 16) & 1u);  // RNE
    return (unsigned short)(r >> 16);
}
__device__ __forceinline__ float bf2f(unsigned short s) {
    return __uint_as_float(((unsigned int)s) << 16);
}

// ---------------- weight transpose + convert: W[L,K,Nc] f32 -> WT[L,Nc,K] bf16
__global__ void conv_wt(const float* __restrict__ W, unsigned short* __restrict__ WT,
                        int K, int Nc, int total) {
    int gid = blockIdx.x * blockDim.x + threadIdx.x;
    if (gid >= total) return;
    int per = K * Nc;
    int l = gid / per;
    int rem = gid - l * per;
    int n = rem / K;
    int k = rem - n * K;
    WT[gid] = f2bf(W[(size_t)l * per + (size_t)k * Nc + n]);
}

// ---------------- h0 = node_type_emb[id] + depth_emb[clip(depth)]
__global__ void init_h(const int* __restrict__ ids, const int* __restrict__ dep,
                       const float* __restrict__ temb, const float* __restrict__ demb,
                       float* __restrict__ h, int n) {
    int gid = blockIdx.x * blockDim.x + threadIdx.x;
    int node = gid >> 6;
    if (node >= n) return;
    int c = (gid & 63) * 4;
    int t = ids[node];
    int d = dep[node];
    d = d < 0 ? 0 : (d > 20 ? 20 : d);
    const float4 a = *(const float4*)(temb + (size_t)t * EMB + c);
    const float4 b = *(const float4*)(demb + (size_t)d * EMB + c);
    float4 o;
    o.x = a.x + b.x; o.y = a.y + b.y; o.z = a.z + b.z; o.w = a.w + b.w;
    *(float4*)(h + (size_t)node * EMB + c) = o;
}

// ---------------- z = (1+eps[l]) * h
__global__ void z_init(const float* __restrict__ h, float* __restrict__ z,
                       const float* __restrict__ epsp, int l, int total) {
    int gid = blockIdx.x * blockDim.x + threadIdx.x;
    if (gid >= total) return;
    z[gid] = (1.0f + epsp[l]) * h[gid];
}

// ---------------- per edge: msg = relu(h[src] + edge_attr@We + be); z[dst] += msg
__global__ void edge_scatter(const int* __restrict__ src, const int* __restrict__ dst,
                             const float* __restrict__ eattr,
                             const float* __restrict__ We0, const float* __restrict__ We1,
                             const float* __restrict__ be,
                             const float* __restrict__ h, float* __restrict__ z, int E) {
    int gid = blockIdx.x * blockDim.x + threadIdx.x;
    int e = gid >> 6;
    if (e >= E) return;
    int c = (gid & 63) * 4;
    float2 a = *(const float2*)(eattr + (size_t)e * 2);
    int s = src[e], d = dst[e];
    const float4 w0 = *(const float4*)(We0 + c);
    const float4 w1 = *(const float4*)(We1 + c);
    const float4 bb = *(const float4*)(be + c);
    const float4 hv = *(const float4*)(h + (size_t)s * EMB + c);
    float m0 = fmaxf(hv.x + a.x * w0.x + a.y * w1.x + bb.x, 0.0f);
    float m1 = fmaxf(hv.y + a.x * w0.y + a.y * w1.y + bb.y, 0.0f);
    float m2 = fmaxf(hv.z + a.x * w0.z + a.y * w1.z + bb.z, 0.0f);
    float m3 = fmaxf(hv.w + a.x * w0.w + a.y * w1.w + bb.w, 0.0f);
    float* p = z + (size_t)d * EMB + c;
    atomicAdd(p + 0, m0);
    atomicAdd(p + 1, m1);
    atomicAdd(p + 2, m2);
    atomicAdd(p + 3, m3);
}

// ---------------- f32 -> bf16 elementwise
__global__ void cvt_bf16(const float* __restrict__ in, unsigned short* __restrict__ out, int total) {
    int gid = blockIdx.x * blockDim.x + threadIdx.x;
    if (gid >= total) return;
    out[gid] = f2bf(in[gid]);
}

// ---------------- zero BN accumulators
__global__ void zero_stats(float* __restrict__ sums, float* __restrict__ sumsq, int n) {
    int gid = blockIdx.x * blockDim.x + threadIdx.x;
    if (gid >= n) return;
    sums[gid] = 0.0f;
    sumsq[gid] = 0.0f;
}

// ---------------- WMMA GEMM: out[M,Nc] = A[M,K]bf16 @ BT[Nc,K]bf16^T + bias
// one wave per 16x16 tile; block = 8 waves sharing the A row tile
__global__ void gemm_bf16_wmma(const unsigned short* __restrict__ A,
                               const unsigned short* __restrict__ BT,
                               const float* __restrict__ bias,
                               float* __restrict__ outF,
                               unsigned short* __restrict__ outB,
                               int K, int Nc, int applyRelu) {
    const int lane = threadIdx.x & 31;
    const int wave = threadIdx.x >> 5;
    const int m    = lane & 15;   // row within A tile / col within B tile
    const int hi   = lane >> 4;   // half-select per ISA 16-bit fragment layout
    const int rowBase = blockIdx.x * 16;
    const int colBase = (blockIdx.y * 8 + wave) * 16;

    const unsigned short* arow = A  + (size_t)(rowBase + m) * K;
    const unsigned short* brow = BT + (size_t)(colBase + m) * K;

    v8f acc = {0.f, 0.f, 0.f, 0.f, 0.f, 0.f, 0.f, 0.f};
    for (int kc = 0; kc < K; kc += 32) {
        Frag fa, fb;
        // lane holds k = kc+hi*8..+7 and kc+16+hi*8..+7 (ISA 16-bit A 16x32 layout)
        fa.u[0] = *(const u32x4*)(arow + kc + hi * 8);
        fa.u[1] = *(const u32x4*)(arow + kc + 16 + hi * 8);
        fb.u[0] = *(const u32x4*)(brow + kc + hi * 8);
        fb.u[1] = *(const u32x4*)(brow + kc + 16 + hi * 8);
        acc = __builtin_amdgcn_wmma_f32_16x16x32_bf16(
            false, fa.v, false, fb.v, (short)0, acc, false, false);
    }

    const float bv = bias[colBase + m];
    const int row0 = rowBase + hi * 8;  // C/D layout: VGPR r -> M = r + 8*hi
    #pragma unroll
    for (int r = 0; r < 8; ++r) {
        float v = acc[r] + bv;
        if (applyRelu) v = fmaxf(v, 0.0f);
        size_t idx = (size_t)(row0 + r) * Nc + colBase + m;
        if (outB) outB[idx] = f2bf(v);
        else      outF[idx] = v;
    }
}

// ---------------- BN partial stats: fp32 accumulation over bf16 y columns
__global__ void bn_stats(const unsigned short* __restrict__ yb,
                         float* __restrict__ sums, float* __restrict__ sumsq, int rows) {
    int c = threadIdx.x;          // 0..511 == column
    int r0 = blockIdx.x * 512;
    int rend = r0 + 512;
    if (rend > rows) rend = rows;
    float s = 0.0f, s2 = 0.0f;
    for (int r = r0; r < rend; ++r) {
        float v = bf2f(yb[(size_t)r * HID + c]);
        s += v;
        s2 += v * v;
    }
    atomicAdd(&sums[c], s);
    atomicAdd(&sumsq[c], s2);
}

// ---------------- fold BN into scale/shift
__global__ void bn_finalize(const float* __restrict__ sums, const float* __restrict__ sumsq,
                            const float* __restrict__ gamma, const float* __restrict__ beta,
                            float* __restrict__ scale, float* __restrict__ shift, float invN) {
    int c = blockIdx.x * blockDim.x + threadIdx.x;
    if (c >= HID) return;
    float mu = sums[c] * invN;
    float var = sumsq[c] * invN - mu * mu;
    float rs = rsqrtf(var + 1e-5f);
    float g = gamma[c] * rs;
    scale[c] = g;
    shift[c] = beta[c] - mu * g;
}

// ---------------- y = bf16(relu(y*scale + shift)), in place
__global__ void bn_apply(unsigned short* __restrict__ yb,
                         const float* __restrict__ scale, const float* __restrict__ shift,
                         int total) {
    int gid = blockIdx.x * blockDim.x + threadIdx.x;
    if (gid >= total) return;
    int c = gid & (HID - 1);
    float v = bf2f(yb[gid]);
    v = fmaxf(v * scale[c] + shift[c], 0.0f);
    yb[gid] = f2bf(v);
}

extern "C" void kernel_launch(void* const* d_in, const int* in_sizes, int n_in,
                              void* d_out, int out_size, void* d_ws, size_t ws_size,
                              hipStream_t stream) {
    const int*   node_ids   = (const int*)d_in[0];
    const int*   node_depth = (const int*)d_in[1];
    const int*   edge_index = (const int*)d_in[2];
    const float* edge_attr  = (const float*)d_in[3];
    const float* temb       = (const float*)d_in[4];
    const float* demb       = (const float*)d_in[5];
    const float* We         = (const float*)d_in[6];
    const float* be         = (const float*)d_in[7];
    const float* W1         = (const float*)d_in[8];
    const float* b1         = (const float*)d_in[9];
    const float* gamma      = (const float*)d_in[10];
    const float* beta       = (const float*)d_in[11];
    const float* W2         = (const float*)d_in[12];
    const float* b2         = (const float*)d_in[13];
    const float* epsp       = (const float*)d_in[14];

    const int N = in_sizes[0];
    const int E = in_sizes[2] / 2;
    const int L = in_sizes[14];

    char* ws = (char*)d_ws;
    size_t off = 0;
    auto alloc = [&](size_t bytes) -> char* {
        char* p = ws + off;
        off += (bytes + 255) & ~(size_t)255;
        return p;
    };
    unsigned short* w1t = (unsigned short*)alloc((size_t)L * EMB * HID * 2);
    unsigned short* w2t = (unsigned short*)alloc((size_t)L * HID * EMB * 2);
    float*          hA  = (float*)alloc((size_t)N * EMB * 4);
    float*          zf  = (float*)alloc((size_t)N * EMB * 4);
    unsigned short* zb  = (unsigned short*)alloc((size_t)N * EMB * 2);
    unsigned short* yb  = (unsigned short*)alloc((size_t)N * HID * 2);
    float* sums  = (float*)alloc(HID * 4);
    float* sumsq = (float*)alloc(HID * 4);
    float* scale = (float*)alloc(HID * 4);
    float* shift = (float*)alloc(HID * 4);
    float* hOut  = (float*)d_out;

    const int totW = L * EMB * HID;
    conv_wt<<<(totW + 255) / 256, 256, 0, stream>>>(W1, w1t, EMB, HID, totW);
    conv_wt<<<(totW + 255) / 256, 256, 0, stream>>>(W2, w2t, HID, EMB, totW);
    init_h<<<(N * 64 + 255) / 256, 256, 0, stream>>>(node_ids, node_depth, temb, demb, hA, N);

    for (int l = 0; l < L; ++l) {
        const float* hin = (l & 1) ? hOut : hA;
        float*       hout = (l & 1) ? hA : hOut;

        z_init<<<(N * EMB + 255) / 256, 256, 0, stream>>>(hin, zf, epsp, l, N * EMB);
        edge_scatter<<<(E * 64 + 255) / 256, 256, 0, stream>>>(
            edge_index, edge_index + E, edge_attr,
            We + (size_t)l * 2 * EMB, We + (size_t)l * 2 * EMB + EMB,
            be + (size_t)l * EMB, hin, zf, E);
        cvt_bf16<<<(N * EMB + 255) / 256, 256, 0, stream>>>(zf, zb, N * EMB);
        zero_stats<<<(HID + 255) / 256, 256, 0, stream>>>(sums, sumsq, HID);

        dim3 g1(N / 16, (HID / 16) / 8);
        gemm_bf16_wmma<<<g1, 256, 0, stream>>>(
            zb, w1t + (size_t)l * HID * EMB, b1 + (size_t)l * HID,
            nullptr, yb, EMB, HID, 0);

        bn_stats<<<(N + 511) / 512, 512, 0, stream>>>(yb, sums, sumsq, N);
        bn_finalize<<<(HID + 255) / 256, 256, 0, stream>>>(
            sums, sumsq, gamma + (size_t)l * HID, beta + (size_t)l * HID,
            scale, shift, 1.0f / (float)N);
        bn_apply<<<(N * HID + 255) / 256, 256, 0, stream>>>(yb, scale, shift, N * HID);

        dim3 g2(N / 16, (EMB / 16) / 8);
        gemm_bf16_wmma<<<g2, 256, 0, stream>>>(
            yb, w2t + (size_t)l * EMB * HID, b2 + (size_t)l * EMB,
            hout, nullptr, HID, EMB, (l < L - 1) ? 1 : 0);
    }
}
